// RSKHRNNCell_56126632624462
// MI455X (gfx1250) — matrix operand order repaired and checked by reading, not compiled, not run on hardware
//
#include <hip/hip_runtime.h>
#include <hip/hip_bf16.h>
#include <math.h>

// CDNA5 / gfx1250, wave32. Pre-packed bf16 hi/lo split-precision WMMA GEMM with
// TDM (tensor_load_to_lds) double-buffered LDS staging.

typedef __attribute__((ext_vector_type(16))) __bf16       v16bf;
typedef __attribute__((ext_vector_type(8)))  __bf16       v8bf;
typedef __attribute__((ext_vector_type(4)))  __bf16       v4bf;
typedef __attribute__((ext_vector_type(8)))  float        v8f;
typedef __attribute__((ext_vector_type(4)))  unsigned int v4u;
typedef __attribute__((ext_vector_type(8)))  int          v8i;
typedef __attribute__((ext_vector_type(4)))  int          v4i;

#define WAVE 32
#define EPSQ 1e-8f

#if __has_builtin(__builtin_amdgcn_tensor_load_to_lds) && __has_builtin(__builtin_amdgcn_s_wait_tensorcnt)
#define TDM_OK 1
#endif

// LDS tile geometry: rows of 32 bf16 (64B) padded to 40 bf16 (80B) so the
// 16-lane x 16B fragment reads hit distinct bank groups (20 banks/row step).
#define LDS_ROW 40

__device__ __forceinline__ v8f wmma_bf16(v16bf a, v16bf b, v8f c) {
  return __builtin_amdgcn_wmma_f32_16x16x32_bf16(false, a, false, b, (short)0, c, false, false);
}

#if defined(TDM_OK)
// TDM: load a 2D tile of `rows` x 32 bf16 (row stride K elements) into LDS at
// lds_off, padding each 64B row with 16B (-> 80B LDS stride). D# layout per
// CDNA5 ISA.
__device__ __forceinline__ void tdm_load_tile(const __bf16* gsrc, unsigned lds_off,
                                              unsigned rows, unsigned K) {
  unsigned long long ga = (unsigned long long)(uintptr_t)gsrc;
  v4u g0;
  g0[0] = 1u;                                   // count=1, user mode
  g0[1] = lds_off;                              // lds_addr
  g0[2] = (unsigned)ga;                         // global_addr[31:0]
  g0[3] = (unsigned)(ga >> 32) | (2u << 30);    // global_addr[56:32] | type=2
  v8i g1;
  g1[0] = (1 << 16)      // data_size = 2B
        | (1 << 20)      // pad_enable
        | (3 << 22)      // pad_interval: 16 dwords (64B)
        | (3 << 25);     // pad_amount: 4 dwords (16B)
  g1[1] = (int)(32u << 16);                     // tensor_dim0[15:0]=32 (== tile, interior tiles)
  g1[2] = (int)((rows & 0xFFFFu) << 16);        // tensor_dim0[31:16]=0 | tensor_dim1[15:0]=rows
  g1[3] = (int)(32u << 16);                     // tensor_dim1[31:16]=0 | tile_dim0=32
  g1[4] = (int)(rows & 0xFFFFu);                // tile_dim1=rows, tile_dim2=0
  g1[5] = (int)K;                               // tensor_dim0_stride[31:0]
  g1[6] = 0;
  g1[7] = 0;
  v4i z4 = {0, 0, 0, 0};
#if __clang_major__ >= 23
  v8i z8 = {0, 0, 0, 0, 0, 0, 0, 0};
  __builtin_amdgcn_tensor_load_to_lds(g0, g1, z4, z4, z8, 0);
#else
  __builtin_amdgcn_tensor_load_to_lds(g0, g1, z4, z4, 0);
#endif
}
#endif

// A fragment (16-bit A 16x32 ISA layout): lane half selects K sub-ranges.
__device__ __forceinline__ v16bf ld_afrag(const __bf16* plane, int row, int half) {
  const __bf16* p = plane + row * LDS_ROW;
  v8bf p0 = *(const v8bf*)(p + 8 * half);        // K 0..7  / 8..15
  v8bf p1 = *(const v8bf*)(p + 16 + 8 * half);   // K 16..23 / 24..31
  return __builtin_shufflevector(p0, p1, 0,1,2,3,4,5,6,7,8,9,10,11,12,13,14,15);
}
// B fragment (B 32x16 ISA layout): lane half selects K sub-ranges, matching A.
__device__ __forceinline__ v16bf ld_bfrag(const __bf16* plane, int col, int half) {
  const __bf16* p = plane + col * LDS_ROW;
  v8bf p0 = *(const v8bf*)(p + 8 * half);
  v8bf p1 = *(const v8bf*)(p + 16 + 8 * half);
  return __builtin_shufflevector(p0, p1, 0,1,2,3,4,5,6,7,8,9,10,11,12,13,14,15);
}

// C[m,n] = sum_k A[m,k]*W[n,k] + bias[n], A/W given as bf16 hi/lo planes (row-major, K contig).
// WG tile 128x128, 8 waves (2 row x 4 col), wave tile 64x32 = 4x2 WMMA tiles.
// MODE 0: out[m*N+n] = C. MODE 1: g=sigmoid(C); out4 = st + g*(ns-st).
template <int MODE>
__global__ __launch_bounds__(256) void gemm_wmma_kernel(
    const __bf16* __restrict__ Ahi, const __bf16* __restrict__ Alo,
    const __bf16* __restrict__ Bhi, const __bf16* __restrict__ Blo,
    const float* __restrict__ bias, float* __restrict__ out,
    const float4* __restrict__ state4, const float4* __restrict__ next4,
    int M, int N, int K) {
  __shared__ __align__(16) __bf16 smem[2][4][128 * LDS_ROW];  // [buf][Ahi,Alo,Bhi,Blo]

  const int tid  = threadIdx.x;
  const int lane = tid & (WAVE - 1);
  const int wv   = tid >> 5;
  const int wrow = wv >> 2;      // 0..1
  const int wcol = wv & 3;       // 0..3
  const int half = lane >> 4;
  const int l15  = lane & 15;

  const int rowBase = blockIdx.y * 128;
  const int colBase = blockIdx.x * 128;
  const int nk = K / 32;

  v8f acc[4][2];
#pragma unroll
  for (int i = 0; i < 4; ++i)
#pragma unroll
    for (int j = 0; j < 2; ++j)
#pragma unroll
      for (int e = 0; e < 8; ++e) acc[i][j][e] = 0.0f;

#if defined(TDM_OK)
  const __bf16* gsrc[4] = {Ahi + (size_t)rowBase * K, Alo + (size_t)rowBase * K,
                           Bhi + (size_t)colBase * K, Blo + (size_t)colBase * K};
  if (wv == 0) {
#pragma unroll
    for (int p = 0; p < 4; ++p)
      tdm_load_tile(gsrc[p], (unsigned)(uintptr_t)&smem[0][p][0], 128, (unsigned)K);
  }
  for (int kt = 0; kt < nk; ++kt) {
    const int cur = kt & 1;
    if (wv == 0) __builtin_amdgcn_s_wait_tensorcnt(0);
    __syncthreads();
    if (wv == 0 && kt + 1 < nk) {
#pragma unroll
      for (int p = 0; p < 4; ++p)
        tdm_load_tile(gsrc[p] + (size_t)(kt + 1) * 32,
                      (unsigned)(uintptr_t)&smem[cur ^ 1][p][0], 128, (unsigned)K);
    }
#else
  for (int kt = 0; kt < nk; ++kt) {
    const int cur = 0;
    __syncthreads();
    {  // cooperative copy: 4 planes x 128 rows x 4 chunks of 16B
      const __bf16* srcp[4] = {Ahi + (size_t)rowBase * K, Alo + (size_t)rowBase * K,
                               Bhi + (size_t)colBase * K, Blo + (size_t)colBase * K};
      for (int c = tid; c < 2048; c += 256) {
        const int p = c >> 9, rem = c & 511, r = rem >> 2, ch = rem & 3;
        const uint4 v = *(const uint4*)(srcp[p] + (size_t)r * K + kt * 32 + ch * 8);
        *(uint4*)&smem[0][p][r * LDS_ROW + ch * 8] = v;
      }
    }
    __syncthreads();
#endif
    // ---- fragments from LDS ----
    v16bf afh[4], afl[4], bfh[2], bfl[2];
#pragma unroll
    for (int i = 0; i < 4; ++i) {
      const int r = wrow * 64 + i * 16 + l15;
      afh[i] = ld_afrag(&smem[cur][0][0], r, half);
      afl[i] = ld_afrag(&smem[cur][1][0], r, half);
    }
#pragma unroll
    for (int j = 0; j < 2; ++j) {
      const int cn = wcol * 32 + j * 16 + l15;
      bfh[j] = ld_bfrag(&smem[cur][2][0], cn, half);
      bfl[j] = ld_bfrag(&smem[cur][3][0], cn, half);
    }
    // ---- split-precision accumulate: hi*hi + hi*lo + lo*hi ----
#pragma unroll
    for (int i = 0; i < 4; ++i)
#pragma unroll
      for (int j = 0; j < 2; ++j) {
        acc[i][j] = wmma_bf16(afh[i], bfh[j], acc[i][j]);
        acc[i][j] = wmma_bf16(afh[i], bfl[j], acc[i][j]);
        acc[i][j] = wmma_bf16(afl[i], bfh[j], acc[i][j]);
      }
  }

  // ---- epilogue (C layout: elem e -> row offset e + 8*half, col l15) ----
#pragma unroll
  for (int j = 0; j < 2; ++j) {
    const int col = colBase + wcol * 32 + j * 16 + l15;
    const float bs = bias[col];
#pragma unroll
    for (int i = 0; i < 4; ++i) {
#pragma unroll
      for (int e = 0; e < 8; ++e) {
        const int row = rowBase + wrow * 64 + i * 16 + e + 8 * half;
        const float val = acc[i][j][e] + bs;
        const size_t idx = (size_t)row * N + col;
        if (MODE == 0) {
          out[idx] = val;
        } else {
          const float g = 1.0f / (1.0f + expf(-val));
          const float4 st = state4[idx];
          const float4 ns = next4[idx];
          float4 r;
          r.x = st.x + g * (ns.x - st.x);
          r.y = st.y + g * (ns.y - st.y);
          r.z = st.z + g * (ns.z - st.z);
          r.w = st.w + g * (ns.w - st.w);
          ((float4*)out)[idx] = r;
        }
      }
    }
  }
}

// f32 -> bf16 hi/lo planes, 8 elements per thread.
__global__ __launch_bounds__(256) void pack_kernel(const float* __restrict__ src,
                                                   __bf16* __restrict__ hi,
                                                   __bf16* __restrict__ lo, int n8) {
  const int i = blockIdx.x * blockDim.x + threadIdx.x;
  if (i >= n8) return;
  const float4 a = ((const float4*)src)[2 * i];
  const float4 b = ((const float4*)src)[2 * i + 1];
  float f[8] = {a.x, a.y, a.z, a.w, b.x, b.y, b.z, b.w};
  v8bf h, l;
#pragma unroll
  for (int e = 0; e < 8; ++e) {
    __bf16 hh = (__bf16)f[e];
    h[e] = hh;
    l[e] = (__bf16)(f[e] - (float)hh);
  }
  ((v8bf*)hi)[i] = h;
  ((v8bf*)lo)[i] = l;
}

// rotation = exp_map(v[1:]), next = qnormalize(qmul(state, rot)); also emits
// bf16 hi/lo packs of next_state and the drift partial sums (wave32 reduce).
__global__ __launch_bounds__(256) void quat_kernel(
    const float4* __restrict__ v4, const float4* __restrict__ state4,
    float4* __restrict__ next4, __bf16* __restrict__ nhi, __bf16* __restrict__ nlo,
    float* __restrict__ drift_acc, int n) {
  const int i = blockIdx.x * blockDim.x + threadIdx.x;
  float d = 0.0f;
  if (i < n) {
    const float4 vv = v4[i];
    const float vx = vv.y, vy = vv.z, vz = vv.w;
    const float theta = sqrtf(vx * vx + vy * vy + vz * vz) + EPSQ;
    const float sc = sinf(theta) / theta;
    const float rw = cosf(theta);
    const float rx = vx * sc, ry = vy * sc, rz = vz * sc;
    const float4 s = state4[i];
    const float aw = s.x, ax = s.y, ay = s.z, az = s.w;
    const float qw = aw * rw - (ax * rx + ay * ry + az * rz);
    const float qx = aw * rx + rw * ax + (ay * rz - az * ry);
    const float qy = aw * ry + rw * ay + (az * rx - ax * rz);
    const float qz = aw * rz + rw * az + (ax * ry - ay * rx);
    const float nrm = sqrtf(qw * qw + qx * qx + qy * qy + qz * qz) + EPSQ;
    const float inv = 1.0f / nrm;
    const float nw = qw * inv, nx = qx * inv, ny = qy * inv, nz = qz * inv;
    next4[i] = make_float4(nw, nx, ny, nz);
    const float f[4] = {nw, nx, ny, nz};
    v4bf h, l;
#pragma unroll
    for (int e = 0; e < 4; ++e) {
      __bf16 hh = (__bf16)f[e];
      h[e] = hh;
      l[e] = (__bf16)(f[e] - (float)hh);
    }
    ((v4bf*)nhi)[i] = h;
    ((v4bf*)nlo)[i] = l;
    const float nn = sqrtf(nw * nw + nx * nx + ny * ny + nz * nz);
    d = fabsf(nn - 1.0f);
  }
#pragma unroll
  for (int off = 16; off > 0; off >>= 1) d += __shfl_down(d, off, WAVE);
  if ((threadIdx.x & (WAVE - 1)) == 0) atomicAdd(drift_acc, d);
}

__global__ void zero_kernel(float* p) { p[0] = 0.0f; }
__global__ void eta_kernel(const float* __restrict__ acc, float* __restrict__ dst, float scale) {
  dst[0] = acc[0] * scale;
}

extern "C" void kernel_launch(void* const* d_in, const int* in_sizes, int n_in,
                              void* d_out, int out_size, void* d_ws, size_t ws_size,
                              hipStream_t stream) {
  const float* x     = (const float*)d_in[0];   // (B, D)
  const float* state = (const float*)d_in[1];   // (B, Q, 4)
  const float* W_phi = (const float*)d_in[2];   // (D, D)
  const float* b_phi = (const float*)d_in[3];
  const float* W_dde = (const float*)d_in[4];   // (Q, D)
  const float* b_dde = (const float*)d_in[5];

  const int B = 2048, D = 4096, Q = 1024;
  float* out = (float*)d_out;

  // workspace carve-up (all 256B aligned)
  char* w = (char*)d_ws;
  float*  drift = (float*)w;                     w += 256;
  float*  nexts = (float*)w;                     w += (size_t)B * D * 4;   // 32MB f32
  __bf16* x_hi  = (__bf16*)w;                    w += (size_t)B * D * 2;   // 16MB
  __bf16* x_lo  = (__bf16*)w;                    w += (size_t)B * D * 2;
  __bf16* wp_hi = (__bf16*)w;                    w += (size_t)D * D * 2;   // 32MB
  __bf16* wp_lo = (__bf16*)w;                    w += (size_t)D * D * 2;
  __bf16* wd_hi = (__bf16*)w;                    w += (size_t)Q * D * 2;   // 8MB
  __bf16* wd_lo = (__bf16*)w;                    w += (size_t)Q * D * 2;
  __bf16* ns_hi = (__bf16*)w;                    w += (size_t)B * D * 2;
  __bf16* ns_lo = (__bf16*)w;                    w += (size_t)B * D * 2;

  zero_kernel<<<1, 1, 0, stream>>>(drift);

  // pack f32 inputs into bf16 hi/lo planes
  {
    int n8 = B * D / 8;
    pack_kernel<<<(n8 + 255) / 256, 256, 0, stream>>>(x, x_hi, x_lo, n8);
    n8 = D * D / 8;
    pack_kernel<<<(n8 + 255) / 256, 256, 0, stream>>>(W_phi, wp_hi, wp_lo, n8);
    n8 = Q * D / 8;
    pack_kernel<<<(n8 + 255) / 256, 256, 0, stream>>>(W_dde, wd_hi, wd_lo, n8);
  }

  // GEMM1: v = x @ W_phi^T + b_phi -> staged into d_out (B*D floats before eta slot)
  {
    dim3 grid(D / 128, B / 128);
    gemm_wmma_kernel<0><<<grid, 256, 0, stream>>>(x_hi, x_lo, wp_hi, wp_lo, b_phi, out,
                                                  nullptr, nullptr, B, D, D);
  }

  // exp_map + qmul + normalize + drift + next_state bf16 packs
  {
    const int n = B * Q;
    quat_kernel<<<(n + 255) / 256, 256, 0, stream>>>(
        (const float4*)out, (const float4*)state, (float4*)nexts, ns_hi, ns_lo, drift, n);
  }

  // GEMM2: gate = sigmoid(next @ W_dde^T + b_dde); fused blend into d_out
  {
    dim3 grid(Q / 128, B / 128);
    gemm_wmma_kernel<1><<<grid, 256, 0, stream>>>(ns_hi, ns_lo, wd_hi, wd_lo, b_dde, out,
                                                  (const float4*)state, (const float4*)nexts,
                                                  B, Q, D);
  }

  eta_kernel<<<1, 1, 0, stream>>>(drift, out + (size_t)B * D, 1.0f / (float)(B * Q));
}
